// KNNC_75763223102128
// MI455X (gfx1250) — 8-bit matrix operand order repaired and checked by reading, compiled, not run on hardware
//
#include <hip/hip_runtime.h>
#include <stdint.h>

typedef int v8i __attribute__((ext_vector_type(8)));

#define KSEL 16
#define NCLS 100
#define CAP 4096
#define NWAVES 8
#define HPAD 257

__device__ __forceinline__ uint32_t fkey(float f) {
  uint32_t u = __float_as_uint(f);
  uint32_t m = (uint32_t)(((int32_t)u) >> 31) | 0x80000000u;
  return u ^ m;
}

__global__ __launch_bounds__(256) void knnc_topk_mode(const float* __restrict__ dist,
                                                      const int* __restrict__ labels,
                                                      float* __restrict__ out,
                                                      int cols) {
  __shared__ uint32_t hist[NWAVES * HPAD];
  __shared__ unsigned long long buf[CAP];
  __shared__ unsigned long long red_k[256];
  __shared__ int red_p[256];
  __shared__ uint32_t s_cnt;
  __shared__ uint32_t s_mask, s_val;
  __shared__ unsigned long long s_khi;
  __shared__ int s_done;
  __shared__ int s_lab[KSEL];
  __shared__ uint32_t s_win[KSEL];

  const int row = blockIdx.x;
  const int tid = threadIdx.x;
  const int wave = tid >> 5;
  const float* rp = dist + (size_t)row * (size_t)cols;
  const int nvec = cols >> 2;
  const bool vec_ok = ((cols & 3) == 0);

  uint32_t mask = 0, val = 0;
  unsigned long long khi = 0;
  int need = KSEL;
  uint32_t base_below = 0;

  for (int shift = 24; shift >= 0; shift -= 8) {
    for (int i = tid; i < NWAVES * HPAD; i += 256) hist[i] = 0;
    __syncthreads();
    uint32_t* myh = &hist[wave * HPAD];
    if (vec_ok) {
      const float4* rp4 = (const float4*)rp;
      for (int i = tid; i < nvec; i += 256) {
        __builtin_prefetch(&rp4[i + 512], 0, 0);
        float4 v = rp4[i];
        uint32_t k0 = fkey(v.x), k1 = fkey(v.y), k2 = fkey(v.z), k3 = fkey(v.w);
        if ((k0 & mask) == val) atomicAdd(&myh[(k0 >> shift) & 255u], 1u);
        if ((k1 & mask) == val) atomicAdd(&myh[(k1 >> shift) & 255u], 1u);
        if ((k2 & mask) == val) atomicAdd(&myh[(k2 >> shift) & 255u], 1u);
        if ((k3 & mask) == val) atomicAdd(&myh[(k3 >> shift) & 255u], 1u);
      }
    } else {
      for (int i = tid; i < cols; i += 256) {
        uint32_t k0 = fkey(rp[i]);
        if ((k0 & mask) == val) atomicAdd(&myh[(k0 >> shift) & 255u], 1u);
      }
    }
    __syncthreads();
    {
      uint32_t tot = 0;
      for (int w = 0; w < NWAVES; ++w) tot += hist[w * HPAD + tid];
      hist[tid] = tot;
    }
    __syncthreads();
    if (tid == 0) {
      uint32_t cum = 0;
      int b = 0;
      for (; b < 255; ++b) {
        uint32_t h = hist[b];
        if (cum + h >= (uint32_t)need) break;
        cum += h;
      }
      uint32_t total_upto = base_below + cum + hist[b];
      need -= (int)cum;
      base_below += cum;
      s_mask = mask | (0xFFu << shift);
      s_val = val | ((uint32_t)b << shift);
      s_khi = (unsigned long long)val + ((unsigned long long)(b + 1) << shift);
      s_done = (total_upto <= (uint32_t)CAP) ? 1 : 0;
    }
    __syncthreads();
    mask = s_mask;
    val = s_val;
    khi = s_khi;
    if (s_done) break;
  }

  if (tid == 0) s_cnt = 0;
  __syncthreads();
  if (vec_ok) {
    const float4* rp4 = (const float4*)rp;
    for (int i = tid; i < nvec; i += 256) {
      float4 v = rp4[i];
      uint32_t kk[4] = {fkey(v.x), fkey(v.y), fkey(v.z), fkey(v.w)};
#pragma unroll
      for (int j = 0; j < 4; ++j) {
        if ((unsigned long long)kk[j] < khi) {
          uint32_t p = atomicAdd(&s_cnt, 1u);
          if (p < CAP) buf[p] = ((unsigned long long)kk[j] << 32) | (uint32_t)(i * 4 + j);
        }
      }
    }
  } else {
    for (int i = tid; i < cols; i += 256) {
      uint32_t k0 = fkey(rp[i]);
      if ((unsigned long long)k0 < khi) {
        uint32_t p = atomicAdd(&s_cnt, 1u);
        if (p < CAP) buf[p] = ((unsigned long long)k0 << 32) | (uint32_t)i;
      }
    }
  }
  __syncthreads();
  const int cnt = (int)(s_cnt < (uint32_t)CAP ? s_cnt : (uint32_t)CAP);

  for (int r = 0; r < KSEL; ++r) {
    unsigned long long m = ~0ull;
    int mp = -1;
    for (int p = tid; p < cnt; p += 256) {
      unsigned long long x = buf[p];
      if (x < m) { m = x; mp = p; }
    }
    red_k[tid] = m;
    red_p[tid] = mp;
    __syncthreads();
    for (int s = 128; s > 0; s >>= 1) {
      if (tid < s && red_k[tid + s] < red_k[tid]) {
        red_k[tid] = red_k[tid + s];
        red_p[tid] = red_p[tid + s];
      }
      __syncthreads();
    }
    if (tid == 0) {
      s_win[r] = (uint32_t)red_k[0];
      int wp = red_p[0];
      if (wp >= 0) buf[wp] = ~0ull;
    }
    __syncthreads();
  }

  if (tid < KSEL) {
    uint32_t wi = s_win[tid];
    s_lab[tid] = (wi < (uint32_t)cols) ? labels[wi] : 0;
  }
  __syncthreads();

  if (tid < 32) {
    const int lane = tid;
    const int n = lane & 15;
    const int half = lane >> 4;
    const int lab = s_lab[n];
    v8i A0 = {}, A1 = {}, B0 = {}, B1 = {};
#pragma unroll
    for (int v = 0; v < 8; ++v) {
      uint32_t wa0 = 0, wa1 = 0, wb0 = 0, wb1 = 0;
#pragma unroll
      for (int b = 0; b < 4; ++b) {
        const int ka = ((v >> 1) << 4) + (half << 3) + ((v & 1) << 2) + b;
        const int kb = ((v >> 1) << 4) + (half << 3) + ((v & 1) << 2) + b;
        wa0 |= (uint32_t)(lab == ka) << (8 * b);
        wa1 |= (uint32_t)(lab == ka + 64) << (8 * b);
        wb0 |= (uint32_t)(lab == kb) << (8 * b);
        wb1 |= (uint32_t)(lab == kb + 64) << (8 * b);
      }
      A0[v] = (int)wa0; A1[v] = (int)wa1; B0[v] = (int)wb0; B1[v] = (int)wb1;
    }
    v8i C = {};
    C = __builtin_amdgcn_wmma_i32_16x16x64_iu8(false, A0, false, B0, C, false, false);
    C = __builtin_amdgcn_wmma_i32_16x16x64_iu8(false, A1, false, B1, C, false, false);
    int colsum = 0;
#pragma unroll
    for (int v = 0; v < 8; ++v) colsum += C[v];
    colsum += __shfl_xor(colsum, 16, 32);
    int score = colsum * (NCLS + 1) - lab;
    int blab = lab;
#pragma unroll
    for (int off = 8; off >= 1; off >>= 1) {
      int os = __shfl_xor(score, off, 32);
      int ol = __shfl_xor(blab, off, 32);
      if (os > score) { score = os; blab = ol; }
    }
    if (lane == 0) out[row] = (float)blab;
  }
}

extern "C" void kernel_launch(void* const* d_in, const int* in_sizes, int n_in,
                              void* d_out, int out_size, void* d_ws, size_t ws_size,
                              hipStream_t stream) {
  const float* dist = (const float*)d_in[0];
  const int* labels = (const int*)d_in[1];
  float* out = (float*)d_out;
  const int cols = in_sizes[1];
  const int rows = in_sizes[0] / cols;
  knnc_topk_mode<<<rows, 256, 0, stream>>>(dist, labels, out, cols);
}
